// Actor_IntentionEncoder_80169859547828
// MI455X (gfx1250) — matrix operand order repaired and checked by reading, not compiled, not run on hardware
//
#include <hip/hip_runtime.h>
#include <hip/hip_bf16.h>
#include <math.h>

typedef _Float16 half_t;
typedef __attribute__((ext_vector_type(16))) _Float16 v16h;
typedef __attribute__((ext_vector_type(8)))  float    v8f;

#define DS_ 32
#define DO_ 32
#define D_  64
#define H_  128
#define B_  512
#define A_  16
#define N_  8192   // B_*A_

// ---------------------------------------------------------------------------
// WMMA core (v_wmma_f32_16x16x32_f16, wave32)
// ---------------------------------------------------------------------------
__device__ __forceinline__ v8f wmma_f16(v16h a, v16h b, v8f c) {
  return __builtin_amdgcn_wmma_f32_16x16x32_f16(
      false, a, false, b, (short)0, c, false, false);
}

// ---------------------------------------------------------------------------
// Swizzled operand layouts: each lane's 16-half fragment is CONTIGUOUS.
// A-ready (16 rows x K): frag base = (kc*32 + lane)*16 ; element (m,k):
__device__ __forceinline__ int a_ready_off(int m, int k) {
  int kc = k >> 5, kl = k & 31;
  int g  = (kl >> 3) & 1;
  int j  = (kl & 7) + ((kl & 16) ? 8 : 0);
  return ((kc * 32) + (g * 16) + m) * 16 + j;      // linear in m: +16 per row
}
__device__ __forceinline__ v16h load_a_ready(const half_t* blk, int kc, int lane) {
  return *(const v16h*)(blk + (kc * 32 + lane) * 16);
}
// B-ready (K x 128 block): frag base = ((kc*8 + w)*32 + lane)*16
__device__ __forceinline__ v16h load_b_ready(const half_t* blk, int w, int kc, int lane) {
  return *(const v16h*)(blk + ((kc * 8 + w) * 32 + lane) * 16);
}

// D (16x16 f32): VGPR v -> M = v + (lane>=16 ? 8:0), N = lane&15
// Store bias+relu result in A-ready order: one base addr + imm offsets (v*16)
__device__ __forceinline__ void store_relu_aready(half_t* blk, int lane, v8f c,
                                                  const float* bias, int colbase) {
  int n = lane & 15, mo = (lane & 16) ? 8 : 0;
  float b = bias[colbase + n];
  half_t* p = blk + a_ready_off(mo, colbase + n);
#pragma unroll
  for (int v = 0; v < 8; ++v) {
    float x = c[v] + b;
    x = x > 0.f ? x : 0.f;
    p[v * 16] = (half_t)x;
  }
}

// ---------------------------------------------------------------------------
// prep: f32 row-major [K, Ntot] weights -> f16 B-ready 128-col blocks
// blkshift = log2(K*128) (all cases are powers of two; avoids int division)
// ---------------------------------------------------------------------------
__global__ __launch_bounds__(256) void k_cvt_bready(const float* __restrict__ src,
                                                    half_t* __restrict__ dst,
                                                    int total, int Ntot, int blkshift) {
  int i = blockIdx.x * 256 + threadIdx.x;
  if (i >= total) return;
  int b = i >> blkshift, il = i & ((1 << blkshift) - 1);
  int j = il & 15, lane = (il >> 4) & 31, w = (il >> 9) & 7, kc = il >> 12;
  int g = lane >> 4, n = lane & 15;
  int k = kc * 32 + g * 8 + ((j < 8) ? j : 16 + (j - 8));
  int c = b * 128 + w * 16 + n;
  dst[i] = (half_t)src[k * Ntot + c];
}

// build x: f32 row-major [N,64] + f16 A-ready blocks [N/16][1024]
__global__ __launch_bounds__(256) void k_build_x(const float* __restrict__ obs,
                                                 const float* __restrict__ latent,
                                                 float* __restrict__ xf,
                                                 half_t* __restrict__ xr) {
  int i = blockIdx.x * 256 + threadIdx.x;          // over N_*D_
  if (i >= N_ * D_) return;
  {
    int n = i >> 6, c = i & 63;
    float v = (c < DS_) ? obs[(n & (B_ - 1)) * DS_ + c] : latent[n * DO_ + (c - DS_)];
    xf[i] = v;
  }
  {
    int rblk = i >> 10, il = i & 1023;
    int j = il & 15, lane = (il >> 4) & 31, kc = il >> 9;
    int m = lane & 15, kb = (lane & 16) ? 8 : 0;
    int k = kc * 32 + kb + ((j < 8) ? j : 16 + (j - 8));
    int row = rblk * 16 + m;
    float v = (k < DS_) ? obs[(row & (B_ - 1)) * DS_ + k] : latent[row * DO_ + (k - DS_)];
    xr[i] = (half_t)v;
  }
}

// ---------------------------------------------------------------------------
// hypernet + einsum + tanh (dominant kernel; no barriers in d-loop)
// ---------------------------------------------------------------------------
__global__ __launch_bounds__(256) void k_hyper(const half_t* __restrict__ xr,
                                               const float* __restrict__ xf,
                                               const half_t* __restrict__ hw1r,
                                               const float* __restrict__ hb1,
                                               const half_t* __restrict__ hw2r,
                                               const float* __restrict__ hb2,
                                               float* __restrict__ embf,
                                               half_t* __restrict__ embh) {
  __shared__ __align__(16) float  xT_sh[64 * 16];
  __shared__ __align__(32) half_t h_sh[16 * 128];

  int tid = threadIdx.x, wave = tid >> 5, lane = tid & 31;
  int rblk = blockIdx.x, row0 = rblk * 16, colbase = wave * 16;
  int n = lane & 15, mo = (lane & 16) ? 8 : 0;

  for (int i = tid; i < 16 * 64; i += 256) {       // x transpose for broadcast reads
    int r = i >> 6, c = i & 63;
    xT_sh[c * 16 + r] = xf[(row0 + r) * 64 + c];
  }

  // layer1: h = relu(x @ hw1 + hb1), K=64 ; A and B both direct from global
  const half_t* xblk = xr + rblk * 1024;
  {
    v8f c = {};
    c = wmma_f16(load_a_ready(xblk, 0, lane), load_b_ready(hw1r, wave, 0, lane), c);
    c = wmma_f16(load_a_ready(xblk, 1, lane), load_b_ready(hw1r, wave, 1, lane), c);
    store_relu_aready(h_sh, lane, c, hb1, colbase);
  }
  __syncthreads();

  v16h a0 = load_a_ready(h_sh, 0, lane);           // loop-invariant A fragments
  v16h a1 = load_a_ready(h_sh, 1, lane);
  v16h a2 = load_a_ready(h_sh, 2, lane);
  v16h a3 = load_a_ready(h_sh, 3, lane);

  v8f emb = {};
  for (int d = 0; d < 64; ++d) {
    const half_t* wblk = hw2r + d * (128 * 128);   // B-ready block, L2-resident
    if (d + 1 < 64)                                // global_prefetch next block
      __builtin_prefetch(wblk + 128 * 128 + ((wave * 4) * 32 + lane) * 16, 0, 1);
    v8f c = {};
    c = wmma_f16(a0, load_b_ready(wblk, wave, 0, lane), c);
    c = wmma_f16(a1, load_b_ready(wblk, wave, 1, lane), c);
    c = wmma_f16(a2, load_b_ready(wblk, wave, 2, lane), c);
    c = wmma_f16(a3, load_b_ready(wblk, wave, 3, lane), c);
    float bias = hb2[d * 128 + colbase + n];
    float xs[8];                                   // 2 x ds_load_b128 broadcast
    *(float4*)(xs)     = *(const float4*)(xT_sh + d * 16 + mo);
    *(float4*)(xs + 4) = *(const float4*)(xT_sh + d * 16 + mo + 4);
#pragma unroll
    for (int v = 0; v < 8; ++v) {
      float g = c[v] + bias;
      g = g > 0.f ? g : 0.f;
      emb[v] += xs[v] * g;
    }
  }

  half_t* ep16 = embh + rblk * 2048 + a_ready_off(mo, colbase + n);
  float*  ep32 = embf + (row0 + mo) * 128 + colbase + n;
#pragma unroll
  for (int v = 0; v < 8; ++v) {
    float e = tanhf(emb[v]);
    ep32[v * 128] = e;
    ep16[v * 16]  = (half_t)e;
  }
}

// ---------------------------------------------------------------------------
// emb_mean over reshape groups of A=16 -> A-ready blocks (32 blocks of 16 rows)
// ---------------------------------------------------------------------------
__global__ __launch_bounds__(128) void k_mean(const float* __restrict__ embf,
                                              half_t* __restrict__ meanh) {
  int b = blockIdx.x, hh = threadIdx.x;
  float s = 0.f;
#pragma unroll
  for (int a = 0; a < A_; ++a) s += embf[(b * A_ + a) * H_ + hh];
  meanh[(b >> 4) * 2048 + a_ready_off(b & 15, hh)] = (half_t)(s * (1.f / A_));
}

// ---------------------------------------------------------------------------
// vals = relu(relu(emb@vw1+vb1)@vw2+vb2)
// ---------------------------------------------------------------------------
__global__ __launch_bounds__(256) void k_vals(const half_t* __restrict__ embh,
                                              const half_t* __restrict__ vw1r,
                                              const float* __restrict__ vb1,
                                              const half_t* __restrict__ vw2r,
                                              const float* __restrict__ vb2,
                                              float* __restrict__ vals) {
  __shared__ __align__(32) half_t mid_sh[16 * 128];
  int tid = threadIdx.x, wave = tid >> 5, lane = tid & 31;
  int rblk = blockIdx.x, row0 = rblk * 16, colbase = wave * 16;

  const half_t* ablk = embh + rblk * 2048;
  v8f c = {};
#pragma unroll
  for (int kc = 0; kc < 4; ++kc)
    c = wmma_f16(load_a_ready(ablk, kc, lane), load_b_ready(vw1r, wave, kc, lane), c);
  store_relu_aready(mid_sh, lane, c, vb1, colbase);
  __syncthreads();

  v8f c2 = {};
#pragma unroll
  for (int kc = 0; kc < 4; ++kc)
    c2 = wmma_f16(load_a_ready(mid_sh, kc, lane), load_b_ready(vw2r, wave, kc, lane), c2);
  int n = lane & 15, mo = (lane & 16) ? 8 : 0;
  float b = vb2[colbase + n];
  float* vp = vals + (row0 + mo) * 128 + colbase + n;
#pragma unroll
  for (int v = 0; v < 8; ++v) {
    float x = c2[v] + b;
    vp[v * 128] = x > 0.f ? x : 0.f;
  }
}

// ---------------------------------------------------------------------------
// attention scores: relu(relu([emb, mean_rep]@aw1+ab1)@aw2+ab2)@aw3 + ab3
// ---------------------------------------------------------------------------
__global__ __launch_bounds__(256) void k_att(const half_t* __restrict__ embh,
                                             const half_t* __restrict__ meanh,
                                             const half_t* __restrict__ aw1r,
                                             const float* __restrict__ ab1,
                                             const half_t* __restrict__ aw2r,
                                             const float* __restrict__ ab2,
                                             const float* __restrict__ aw3,
                                             const float* __restrict__ ab3,
                                             float* __restrict__ scores) {
  __shared__ __align__(32) half_t s1_sh[16 * 128];
  __shared__ __align__(32) half_t s2_sh[16 * 128];      // plain row-major
  int tid = threadIdx.x, wave = tid >> 5, lane = tid & 31;
  int rblk = blockIdx.x, row0 = rblk * 16, colbase = wave * 16;

  const half_t* eblk = embh + rblk * 2048;
  const half_t* mblk = meanh + (rblk & 31) * 2048;      // mean_rep[n] = mean[n % B]

  v8f c1 = {};
#pragma unroll
  for (int kc = 0; kc < 4; ++kc)                        // emb half (aw1 rows 0..127)
    c1 = wmma_f16(load_a_ready(eblk, kc, lane), load_b_ready(aw1r, wave, kc, lane), c1);
#pragma unroll
  for (int kc = 0; kc < 4; ++kc)                        // mean half (aw1 rows 128..255)
    c1 = wmma_f16(load_a_ready(mblk, kc, lane), load_b_ready(aw1r, wave, kc + 4, lane), c1);
  store_relu_aready(s1_sh, lane, c1, ab1, colbase);
  __syncthreads();

  v8f c2 = {};
#pragma unroll
  for (int kc = 0; kc < 4; ++kc)
    c2 = wmma_f16(load_a_ready(s1_sh, kc, lane), load_b_ready(aw2r, wave, kc, lane), c2);
  {
    int n = lane & 15, mo = (lane & 16) ? 8 : 0;
    float b = ab2[colbase + n];
    half_t* sp = s2_sh + mo * 128 + colbase + n;
#pragma unroll
    for (int v = 0; v < 8; ++v) {
      float x = c2[v] + b;
      x = x > 0.f ? x : 0.f;
      sp[v * 128] = (half_t)x;
    }
  }
  __syncthreads();

  if (tid < 16) {                                       // final 128 -> 1 projection
    float s = ab3[0];
    for (int k = 0; k < 128; ++k) s += (float)s2_sh[tid * 128 + k] * aw3[k];
    scores[row0 + tid] = s;
  }
}

// ---------------------------------------------------------------------------
// softmax over groups of A=16 + weighted sum of vals -> out [B,H]
// ---------------------------------------------------------------------------
__global__ __launch_bounds__(128) void k_out(const float* __restrict__ scores,
                                             const float* __restrict__ vals,
                                             float* __restrict__ out) {
  __shared__ float sc[A_];
  int b = blockIdx.x, t = threadIdx.x;
  if (t < A_) sc[t] = scores[b * A_ + t];
  __syncthreads();

  float mx = -INFINITY;
#pragma unroll
  for (int a = 0; a < A_; ++a) mx = fmaxf(mx, sc[a]);
  float w[A_];
  float den = 0.f;
#pragma unroll
  for (int a = 0; a < A_; ++a) { w[a] = expf(sc[a] - mx); den += w[a]; }
  float inv = 1.f / den;

  float s = 0.f;
#pragma unroll
  for (int a = 0; a < A_; ++a) s += (w[a] * inv) * vals[(b * A_ + a) * H_ + t];
  out[b * H_ + t] = s;
}

// ---------------------------------------------------------------------------
extern "C" void kernel_launch(void* const* d_in, const int* in_sizes, int n_in,
                              void* d_out, int out_size, void* d_ws, size_t ws_size,
                              hipStream_t stream) {
  const float* obs    = (const float*)d_in[0];
  const float* latent = (const float*)d_in[1];
  const float* hw1 = (const float*)d_in[2];  const float* hb1 = (const float*)d_in[3];
  const float* hw2 = (const float*)d_in[4];  const float* hb2 = (const float*)d_in[5];
  const float* vw1 = (const float*)d_in[6];  const float* vb1 = (const float*)d_in[7];
  const float* vw2 = (const float*)d_in[8];  const float* vb2 = (const float*)d_in[9];
  const float* aw1 = (const float*)d_in[10]; const float* ab1 = (const float*)d_in[11];
  const float* aw2 = (const float*)d_in[12]; const float* ab2 = (const float*)d_in[13];
  const float* aw3 = (const float*)d_in[14]; const float* ab3 = (const float*)d_in[15];
  float* out = (float*)d_out;

  char* ws = (char*)d_ws;
  size_t off = 0;
  auto alloc = [&](size_t bytes) { char* p = ws + off; off += (bytes + 255) & ~size_t(255); return p; };

  float*  x_f32  = (float*)alloc(N_ * D_ * 4);
  half_t* x_r    = (half_t*)alloc(N_ * D_ * 2);          // A-ready blocks
  half_t* hw1_r  = (half_t*)alloc(D_ * H_ * 2);          // B-ready
  half_t* hw2_r  = (half_t*)alloc(H_ * D_ * H_ * 2);     // 64 B-ready blocks
  half_t* vw1_r  = (half_t*)alloc(H_ * H_ * 2);
  half_t* vw2_r  = (half_t*)alloc(H_ * H_ * 2);
  half_t* aw1_r  = (half_t*)alloc(2 * H_ * H_ * 2);
  half_t* aw2_r  = (half_t*)alloc(H_ * H_ * 2);
  float*  emb_f  = (float*)alloc(N_ * H_ * 4);
  half_t* emb_r  = (half_t*)alloc(N_ * H_ * 2);          // A-ready blocks
  float*  vals_f = (float*)alloc(N_ * H_ * 4);
  half_t* mean_r = (half_t*)alloc(B_ * H_ * 2);          // A-ready blocks
  float*  scores = (float*)alloc(N_ * 4);
  (void)ws_size; (void)in_sizes; (void)n_in; (void)out_size;

  // prep (one-time swizzles); blkshift = log2(K*128)
  k_build_x<<<(N_ * D_ + 255) / 256, 256, 0, stream>>>(obs, latent, x_f32, x_r);
  k_cvt_bready<<<(D_ * H_ + 255) / 256, 256, 0, stream>>>(hw1, hw1_r, D_ * H_, H_, 13);
  k_cvt_bready<<<(H_ * D_ * H_ + 255) / 256, 256, 0, stream>>>(hw2, hw2_r, H_ * D_ * H_, D_ * H_, 14);
  k_cvt_bready<<<(H_ * H_ + 255) / 256, 256, 0, stream>>>(vw1, vw1_r, H_ * H_, H_, 14);
  k_cvt_bready<<<(H_ * H_ + 255) / 256, 256, 0, stream>>>(vw2, vw2_r, H_ * H_, H_, 14);
  k_cvt_bready<<<(2 * H_ * H_ + 255) / 256, 256, 0, stream>>>(aw1, aw1_r, 2 * H_ * H_, H_, 15);
  k_cvt_bready<<<(H_ * H_ + 255) / 256, 256, 0, stream>>>(aw2, aw2_r, H_ * H_, H_, 14);

  // main pipeline
  k_hyper<<<N_ / 16, 256, 0, stream>>>(x_r, x_f32, hw1_r, hb1, hw2_r, hb2, emb_f, emb_r);
  k_mean<<<B_, 128, 0, stream>>>(emb_f, mean_r);
  k_vals<<<N_ / 16, 256, 0, stream>>>(emb_r, vw1_r, vb1, vw2_r, vb2, vals_f);
  k_att<<<N_ / 16, 256, 0, stream>>>(emb_r, mean_r, aw1_r, ab1, aw2_r, ab2, aw3, ab3, scores);
  k_out<<<B_, 128, 0, stream>>>(scores, vals_f, out);
}
